// TransformerGNN_90933047591017
// MI455X (gfx1250) — matrix operand order repaired and checked by reading, not compiled, not run on hardware
//
#include <hip/hip_runtime.h>
#include <hip/hip_fp16.h>

// ---------------------------------------------------------------------------
// TransformerConv GNN (2 layers) for MI455X / gfx1250.
// Dense projections via v_wmma_f32_16x16x32_f16 (f16 in, f32 accumulate).
// Edge softmax-aggregation via f32 atomics (bandwidth-bound scatter phase).
// ---------------------------------------------------------------------------

typedef __attribute__((ext_vector_type(16))) _Float16 v16h;
typedef __attribute__((ext_vector_type(8)))  _Float16 v8h;
typedef __attribute__((ext_vector_type(8)))  float    v8f;

#define NN 50000
#define NE 800000
// flip(-inf): monotone order-preserving float->uint mapping of -infinity
#define FLIP_NEG_INF 0x007FFFFFu

__device__ __forceinline__ unsigned flipF(float f) {
  unsigned u = __float_as_uint(f);
  return (u & 0x80000000u) ? ~u : (u | 0x80000000u);
}
__device__ __forceinline__ float unflipF(unsigned u) {
  u = (u & 0x80000000u) ? (u & 0x7FFFFFFFu) : ~u;
  return __uint_as_float(u);
}

// ---------------- utility kernels ----------------

__global__ void k_f32_to_f16(const float* __restrict__ src,
                             _Float16* __restrict__ dst, int n) {
  int i = blockIdx.x * blockDim.x + threadIdx.x;
  if (i < n) dst[i] = (_Float16)src[i];
}

// w is [K x F] row-major f32; wt becomes [F x K] row-major f16 (transposed)
__global__ void k_w_to_f16_t(const float* __restrict__ w,
                             _Float16* __restrict__ wt, int K, int F) {
  int i = blockIdx.x * blockDim.x + threadIdx.x;
  if (i >= K * F) return;
  int k = i / F, f = i - k * F;
  wt[(size_t)f * K + k] = (_Float16)w[i];
}

__global__ void k_fill_f32(float* __restrict__ p, int n, float v) {
  int i = blockIdx.x * blockDim.x + threadIdx.x;
  if (i < n) p[i] = v;
}
__global__ void k_fill_u32(unsigned* __restrict__ p, int n, unsigned v) {
  int i = blockIdx.x * blockDim.x + threadIdx.x;
  if (i < n) p[i] = v;
}

// ---------------- WMMA GEMM: C[N x F] = A[N x 128](f16) @ Wt^T + bias ------
// A: row-major [NN x 128] f16.  Wt: [F x 128] f16 (pre-transposed weight).
// One 16x16 output tile per wave; K unrolled in steps of 32.
// A-fragment lane layout (ISA 7.12.2, 16-bit A 16x32): lane m=lane&15,
//   group g=lane>>4: elems 0..7 = K[g*8 .. g*8+7], elems 8..15 = K[g*8+16..g*8+23]
// B-fragment (32x16): lane n=lane&15, group g: elems 0..7 = K[g*8 .. g*8+7], elems 8..15 = K[g*8+16..g*8+23]
template <int F>
__global__ __launch_bounds__(256) void k_gemm_wmma(
    const _Float16* __restrict__ A, const _Float16* __restrict__ Wt,
    const float* __restrict__ bias, float* __restrict__ C) {
  constexpr int K = 128;
  constexpr int tilesN = F / 16;
  constexpr int totalTiles = (NN / 16) * tilesN;
  const int wave = threadIdx.x >> 5;
  const int lane = threadIdx.x & 31;
  const int tile = blockIdx.x * 8 + wave;
  if (tile >= totalTiles) return;  // wave-uniform: EXEC stays all-ones
  const int mt = tile / tilesN, nt = tile - mt * tilesN;
  const int row0 = mt * 16, col0 = nt * 16;
  const int m = lane & 15;            // A row within tile == B col within tile
  const int kbA = (lane >> 4) << 3;   // 0 or 8
  const int kbB = (lane >> 4) << 4;   // 0 or 16

  const _Float16* arow = A + (size_t)(row0 + m) * K;
  const _Float16* brow = Wt + (size_t)(col0 + m) * K;

  v8f acc = {};
#pragma unroll
  for (int k0 = 0; k0 < K; k0 += 32) {
    union { v16h v; v8h h[2]; } ua, ub;
    ua.h[0] = *(const v8h*)(arow + k0 + kbA);        // K = kbA .. kbA+7
    ua.h[1] = *(const v8h*)(arow + k0 + kbA + 16);   // K = kbA+16 .. kbA+23
    ub.h[0] = *(const v8h*)(brow + k0 + kbA);        // K = kbA .. kbA+7
    ub.h[1] = *(const v8h*)(brow + k0 + kbA + 16);   // K = kbA+16 .. kbA+23
    acc = __builtin_amdgcn_wmma_f32_16x16x32_f16(
        /*neg_a=*/false, ua.v, /*neg_b=*/false, ub.v,
        /*c_mod=*/(short)0, acc, /*reuse_a=*/false, /*reuse_b=*/false);
  }

  const float bcol = bias[col0 + m];
  const int mbase = row0 + ((lane >> 4) << 3);  // VGPR r -> row mbase + r
#pragma unroll
  for (int r = 0; r < 8; ++r)
    C[(size_t)(mbase + r) * F + col0 + m] = acc[r] + bcol;
}

// ---------------- edge kernels ----------------

// per-(edge, head): logit = scale * dot(q[dst,h,:], k[src,h,:]); segment max
__global__ void k_edge_logits(const float* __restrict__ q,
                              const float* __restrict__ kk,
                              const int* __restrict__ ei,
                              float* __restrict__ logits,
                              unsigned* __restrict__ mx,
                              int H, int D, float scale) {
  int idx = blockIdx.x * blockDim.x + threadIdx.x;
  if (idx >= NE * H) return;
  int e = idx / H, h = idx - e * H;
  int s = ei[e], d = ei[NE + e];
  const float* qp = q  + (size_t)d * H * D + (size_t)h * D;
  const float* kp = kk + (size_t)s * H * D + (size_t)h * D;
  float acc = 0.f;
  for (int i = 0; i < D; ++i) acc += qp[i] * kp[i];
  acc *= scale;
  logits[idx] = acc;
  atomicMax(mx + (size_t)d * H + h, flipF(acc));
}

// uint(flipped-max) -> float in place; isolated nodes (-inf) -> 0
__global__ void k_max_fin(unsigned* __restrict__ buf, int n) {
  int i = blockIdx.x * blockDim.x + threadIdx.x;
  if (i >= n) return;
  unsigned u = buf[i];
  float v = (u == FLIP_NEG_INF) ? 0.f : unflipF(u);
  ((float*)buf)[i] = v;
}

// per-(edge, head): p = exp(logit - m[dst]); overwrite logits with p; denom +=
__global__ void k_edge_p(float* __restrict__ logits,
                         const float* __restrict__ m,
                         const int* __restrict__ ei,
                         float* __restrict__ den, int H) {
  int idx = blockIdx.x * blockDim.x + threadIdx.x;
  if (idx >= NE * H) return;
  int e = idx / H, h = idx - e * H;
  int d = ei[NE + e];
  float p = expf(logits[idx] - m[(size_t)d * H + h]);
  logits[idx] = p;
  atomicAdd(den + (size_t)d * H + h, p);
}

// per-(edge, feature): agg[dst, r] += p[e,h] * v[src, r]
__global__ void k_edge_agg(const float* __restrict__ p,
                           const float* __restrict__ v,
                           const int* __restrict__ ei,
                           float* __restrict__ agg, int H, int D) {
  int idx = blockIdx.x * blockDim.x + threadIdx.x;
  int HD = H * D;
  if (idx >= NE * HD) return;
  int e = idx / HD, r = idx - e * HD;
  int h = r / D;
  int s = ei[e], d = ei[NE + e];
  atomicAdd(agg + (size_t)d * HD + r,
            p[(size_t)e * H + h] * v[(size_t)s * HD + r]);
}

// layer-1 finalize: h1 = relu(agg/denom' + skip), stored as f16 for layer-2 GEMMs
__global__ void k_fin1(const float* __restrict__ agg,
                       const float* __restrict__ den,
                       const float* __restrict__ skip,
                       _Float16* __restrict__ h1h) {
  int i = blockIdx.x * blockDim.x + threadIdx.x;
  if (i >= NN * 128) return;
  int n = i >> 7, r = i & 127, h = r >> 5;  // H1=4, HID=32
  float dd = den[(size_t)n * 4 + h];
  dd = dd > 0.f ? dd : 1.f;
  float val = agg[i] / dd + skip[i];
  val = val > 0.f ? val : 0.f;  // ReLU between layers
  h1h[i] = (_Float16)val;
}

// layer-2 finalize: out = agg/denom' + skip  (no ReLU)
__global__ void k_fin2(const float* __restrict__ agg,
                       const float* __restrict__ den,
                       const float* __restrict__ skip,
                       float* __restrict__ out) {
  int i = blockIdx.x * blockDim.x + threadIdx.x;
  if (i >= NN * 64) return;
  int n = i >> 6;  // H2=1, OUT=64
  float dd = den[n];
  dd = dd > 0.f ? dd : 1.f;
  out[i] = agg[i] / dd + skip[i];
}

// ---------------------------------------------------------------------------

static inline int cdiv(long long a, int b) { return (int)((a + b - 1) / b); }

extern "C" void kernel_launch(void* const* d_in, const int* in_sizes, int n_in,
                              void* d_out, int out_size, void* d_ws, size_t ws_size,
                              hipStream_t stream) {
  (void)in_sizes; (void)n_in; (void)out_size;

  const float* x  = (const float*)d_in[0];
  const int*   ei = (const int*)d_in[1];
  // weights / biases in setup_inputs order: q1,k1,v1,s1 (128x128), q2,k2,v2,s2 (128x64)
  const float* w[8]; const float* b[8];
  for (int i = 0; i < 8; ++i) { w[i] = (const float*)d_in[2 + 2*i]; b[i] = (const float*)d_in[3 + 2*i]; }

  // ---- workspace carve-out (256B aligned regions) ----
  char* base = (char*)d_ws;
  size_t off = 0;
  auto carve = [&](size_t bytes) -> char* {
    char* p = base + off;
    off += (bytes + 255) & ~(size_t)255;
    return p;
  };
  _Float16* XH  = (_Float16*)carve((size_t)NN * 128 * 2);   // x as f16
  _Float16* H1H = (_Float16*)carve((size_t)NN * 128 * 2);   // relu(h1) as f16
  _Float16* WT1[4]; for (int i = 0; i < 4; ++i) WT1[i] = (_Float16*)carve(128 * 128 * 2);
  _Float16* WT2[4]; for (int i = 0; i < 4; ++i) WT2[i] = (_Float16*)carve(128 * 64 * 2);
  unsigned* MAXB = (unsigned*)carve((size_t)NN * 4 * 4);    // seg-max (flipped uint -> float)
  float*    DENB = (float*)carve((size_t)NN * 4 * 4);       // softmax denominators
  float*    Qb   = (float*)carve((size_t)NN * 128 * 4);     // reused by both layers
  float*    Kb   = (float*)carve((size_t)NN * 128 * 4);
  float*    Vb   = (float*)carve((size_t)NN * 128 * 4);
  float*    Sb   = (float*)carve((size_t)NN * 128 * 4);
  float*    AGG  = (float*)carve((size_t)NN * 128 * 4);
  float*    LOG  = (float*)carve((size_t)NE * 4 * 4);       // per-edge logits -> p
  if (off > ws_size) return;  // insufficient scratch: bail deterministically

  const int T = 256;
  const float sc1 = 0.1767766952966369f;  // 1/sqrt(32)
  const float sc2 = 0.125f;               // 1/sqrt(64)

  // ---- precision conversion ----
  k_f32_to_f16<<<cdiv((long long)NN * 128, T), T, 0, stream>>>(x, XH, NN * 128);
  for (int i = 0; i < 4; ++i)
    k_w_to_f16_t<<<cdiv(128 * 128, T), T, 0, stream>>>(w[i], WT1[i], 128, 128);
  for (int i = 0; i < 4; ++i)
    k_w_to_f16_t<<<cdiv(128 * 64, T), T, 0, stream>>>(w[4 + i], WT2[i], 128, 64);

  // ---- layer 1 dense projections (WMMA) ----
  {
    const int blocks = ((NN / 16) * (128 / 16) + 7) / 8;  // 8 waves/block
    k_gemm_wmma<128><<<blocks, 256, 0, stream>>>(XH, WT1[0], b[0], Qb);
    k_gemm_wmma<128><<<blocks, 256, 0, stream>>>(XH, WT1[1], b[1], Kb);
    k_gemm_wmma<128><<<blocks, 256, 0, stream>>>(XH, WT1[2], b[2], Vb);
    k_gemm_wmma<128><<<blocks, 256, 0, stream>>>(XH, WT1[3], b[3], Sb);
  }

  // ---- layer 1 edge softmax + aggregation ----
  k_fill_u32<<<cdiv((long long)NN * 4, T), T, 0, stream>>>(MAXB, NN * 4, FLIP_NEG_INF);
  k_fill_f32<<<cdiv((long long)NN * 4, T), T, 0, stream>>>(DENB, NN * 4, 0.f);
  k_fill_f32<<<cdiv((long long)NN * 128, T), T, 0, stream>>>(AGG, NN * 128, 0.f);

  k_edge_logits<<<cdiv((long long)NE * 4, T), T, 0, stream>>>(Qb, Kb, ei, LOG, MAXB, 4, 32, sc1);
  k_max_fin<<<cdiv((long long)NN * 4, T), T, 0, stream>>>(MAXB, NN * 4);
  k_edge_p<<<cdiv((long long)NE * 4, T), T, 0, stream>>>(LOG, (const float*)MAXB, ei, DENB, 4);
  k_edge_agg<<<cdiv((long long)NE * 128, T), T, 0, stream>>>(LOG, Vb, ei, AGG, 4, 32);
  k_fin1<<<cdiv((long long)NN * 128, T), T, 0, stream>>>(AGG, DENB, Sb, H1H);

  // ---- layer 2 dense projections (WMMA) ----
  {
    const int blocks = ((NN / 16) * (64 / 16) + 7) / 8;
    k_gemm_wmma<64><<<blocks, 256, 0, stream>>>(H1H, WT2[0], b[4], Qb);
    k_gemm_wmma<64><<<blocks, 256, 0, stream>>>(H1H, WT2[1], b[5], Kb);
    k_gemm_wmma<64><<<blocks, 256, 0, stream>>>(H1H, WT2[2], b[6], Vb);
    k_gemm_wmma<64><<<blocks, 256, 0, stream>>>(H1H, WT2[3], b[7], Sb);
  }

  // ---- layer 2 edge softmax + aggregation ----
  k_fill_u32<<<cdiv(NN, T), T, 0, stream>>>(MAXB, NN, FLIP_NEG_INF);
  k_fill_f32<<<cdiv(NN, T), T, 0, stream>>>(DENB, NN, 0.f);
  k_fill_f32<<<cdiv((long long)NN * 64, T), T, 0, stream>>>(AGG, NN * 64, 0.f);

  k_edge_logits<<<cdiv((long long)NE, T), T, 0, stream>>>(Qb, Kb, ei, LOG, MAXB, 1, 64, sc2);
  k_max_fin<<<cdiv(NN, T), T, 0, stream>>>(MAXB, NN);
  k_edge_p<<<cdiv((long long)NE, T), T, 0, stream>>>(LOG, (const float*)MAXB, ei, DENB, 1);
  k_edge_agg<<<cdiv((long long)NE * 64, T), T, 0, stream>>>(LOG, Vb, ei, AGG, 1, 64);
  k_fin2<<<cdiv((long long)NN * 64, T), T, 0, stream>>>(AGG, DENB, Sb, (float*)d_out);
}
